// YInducer_3659312136574
// MI455X (gfx1250) — matrix operand order repaired and checked by reading, not compiled, not run on hardware
//
#include <hip/hip_runtime.h>
#include <hip/hip_bf16.h>

// ---------------------------------------------------------------------------
// ON-LSTM LM forward on MI455X (gfx1250), wave32 + WMMA bf16 + TDM staging.
// All GEMMs use v_wmma_f32_16x16x32_bf16. Weights converted once per call to
// bf16 layout Wt[out][in] so A/B WMMA fragments are contiguous 16B loads.
// Big mixture GEMM (5120x32000, K=1024) uses tensor_load_to_lds (TDM) to
// stage the shared A block in LDS, 32x32 output tile per wave.
// ---------------------------------------------------------------------------

typedef __attribute__((ext_vector_type(16))) __bf16 v16bf;
typedef __attribute__((ext_vector_type(8)))  __bf16 v8bf;
typedef __attribute__((ext_vector_type(8)))  float  v8f;
typedef __attribute__((ext_vector_type(4)))  unsigned int u32x4;
typedef __attribute__((ext_vector_type(8)))  int i32x8;
typedef __attribute__((ext_vector_type(4)))  int i32x4;

#define EMB  512
#define HID  1024
#define NCH  16
#define KMIX 5
#define VOC  32000
#define BB   8
#define TT   129
#define TL   128           // T-1
#define GATE 4128          // 4*HID + 2*NCH
#define BL   (BB*TL)       // 1024 (b,l) rows
#define MROW (BL*KMIX)     // 5120 rows of mixture GEMM
#define MBLK 32            // A rows staged in LDS per block (64KB)
#define NSPL 5             // N splits of the mixture GEMM

__device__ __forceinline__ v16bf load_fragA(const __bf16* rowp, int k, int half) {
  // A layout (16-bit 16x32): lane row = lane&15; k = 8*half + (e&7) + 16*(e>>3)
  const __bf16* p = rowp + k + half * 8;
  v8bf lo = *(const v8bf*)(p);
  v8bf hi = *(const v8bf*)(p + 16);
  v16bf r;
#pragma unroll
  for (int e = 0; e < 8; ++e) { r[e] = lo[e]; r[8 + e] = hi[e]; }
  return r;
}

__device__ __forceinline__ v16bf load_fragB(const __bf16* rowp, int k, int half) {
  // B layout (16-bit 32x16): lane col = lane&15; k = 8*half + (e&7) + 16*(e>>3)
  const __bf16* p = rowp + k + half * 8;
  v8bf lo = *(const v8bf*)(p);
  v8bf hi = *(const v8bf*)(p + 16);
  v16bf r;
#pragma unroll
  for (int e = 0; e < 8; ++e) { r[e] = lo[e]; r[8 + e] = hi[e]; }
  return r;
}

__device__ __forceinline__ v8f wmma_bf16(v16bf a, v16bf b, v8f c) {
  return __builtin_amdgcn_wmma_f32_16x16x32_bf16(false, a, false, b, (short)0, c,
                                                 false, false);
}

// ---------------------------------------------------------------------------
// Prep kernels
// ---------------------------------------------------------------------------

__global__ void k_cvt(const float* __restrict__ src, __bf16* __restrict__ dst,
                      size_t n) {
  size_t i = (size_t)blockIdx.x * blockDim.x + threadIdx.x;
  if (i < n) dst[i] = (__bf16)src[i];
}

// Wt[n][k] = (bf16) W[k][n]   (LDS-tiled transpose-convert), block 32x8
__global__ void k_transpose(const float* __restrict__ W, __bf16* __restrict__ Wt,
                            int K, int N) {
  __shared__ float tile[32][33];
  int kb = blockIdx.y * 32, nb = blockIdx.x * 32;
  int tx = threadIdx.x, ty = threadIdx.y;
#pragma unroll
  for (int j = 0; j < 32; j += 8) {
    int k = kb + ty + j, n = nb + tx;
    tile[ty + j][tx] = (k < K && n < N) ? W[(size_t)k * N + n] : 0.f;
  }
  __syncthreads();
#pragma unroll
  for (int j = 0; j < 32; j += 8) {
    int n = nb + ty + j, k = kb + tx;
    if (n < N && k < K) Wt[(size_t)n * K + k] = (__bf16)tile[tx][ty + j];
  }
}

__global__ void k_init(__bf16* hb, float* cb) {
  int i = blockIdx.x * 256 + threadIdx.x;
  if (i < 3 * 2 * 16 * HID) hb[i] = (__bf16)0.f;
  if (i < 3 * 8 * HID) cb[i] = 0.f;
}

// ---------------------------------------------------------------------------
// Xp[t*8+b][g] = emb[tok[b][t]] @ Wx0 + b0   (M=1024, N=4128, K=512) WMMA
// ---------------------------------------------------------------------------
__global__ void k_xproj(const int* __restrict__ tok, const __bf16* __restrict__ embbf,
                        const __bf16* __restrict__ Wx0t, const float* __restrict__ b0,
                        float* __restrict__ Xp) {
  int wid = (blockIdx.x * 256 + threadIdx.x) >> 5;
  int lane = threadIdx.x & 31;
  int mt = wid / 258, nt = wid % 258;
  int half = lane >> 4, r15 = lane & 15;
  int mrow = mt * 16 + r15;            // = t*8+b
  int t = mrow >> 3, b = mrow & 7;
  int tv = tok[b * TT + t];
  const __bf16* arow = embbf + (size_t)tv * EMB;
  int ncol = nt * 16 + r15;
  const __bf16* brow = Wx0t + (size_t)ncol * EMB;
  v8f acc;
  float bv = b0[ncol];
#pragma unroll
  for (int i = 0; i < 8; ++i) acc[i] = bv;
  for (int k = 0; k < EMB; k += 32)
    acc = wmma_bf16(load_fragA(arow, k, half), load_fragB(brow, k, half), acc);
#pragma unroll
  for (int r = 0; r < 8; ++r) {
    int row = mt * 16 + r + 8 * half;
    Xp[(size_t)row * GATE + ncol] = acc[r];
  }
}

// ---------------------------------------------------------------------------
// One ON-LSTM (t, layer) step. grid=32 blocks (32 hidden cols each), 10 waves:
// waves 0..7 -> i/f/o/g tiles, waves 8..9 -> master tiles (redundant per block)
// proj = [Xp | b] + hin@Wxt + hprev@Wht ; then masters softmax/cumsum + cell.
// ---------------------------------------------------------------------------
__global__ void k_lstm_step(int t,
                            const float* __restrict__ Xp,        // layer0 only
                            const __bf16* __restrict__ Wxt,      // null for layer0
                            const __bf16* __restrict__ Wht,
                            const float* __restrict__ bias,      // null for layer0
                            const __bf16* __restrict__ hin,      // prev-layer h (this t)
                            const __bf16* __restrict__ hprev,    // this layer h (t-1)
                            __bf16* __restrict__ hnext,          // this layer h (t)
                            float* __restrict__ cstate,
                            __bf16* __restrict__ h_all,          // layer2 only
                            float* __restrict__ out_df,          // layer1 only
                            float* __restrict__ out_cf) {        // layer1 only
  __shared__ float proj[16][160];
  __shared__ float tfs[8][16], tis[8][16];
  int tid = threadIdx.x, lane = tid & 31, w = tid >> 5;
  int half = lane >> 4, r15 = lane & 15;
  int n0 = blockIdx.x * 32;
  int col;
  if (w < 8) { int g = w >> 1; col = g * HID + n0 + (w & 1) * 16 + r15; }
  else       { col = 4 * HID + (w - 8) * 16 + r15; }

  v8f acc;
  if (Xp) {
#pragma unroll
    for (int r = 0; r < 8; ++r) {
      int row = r + 8 * half;
      acc[r] = (row < 8) ? Xp[((size_t)t * 8 + row) * GATE + col] : 0.f;
    }
  } else {
    float bv = bias[col];
#pragma unroll
    for (int r = 0; r < 8; ++r) acc[r] = bv;
  }
  if (hin) {
    const __bf16* brow = Wxt + (size_t)col * HID;
    const __bf16* arow = hin + (size_t)r15 * HID;
    for (int k = 0; k < HID; k += 32)
      acc = wmma_bf16(load_fragA(arow, k, half), load_fragB(brow, k, half), acc);
  }
  {
    const __bf16* brow = Wht + (size_t)col * HID;
    const __bf16* arow = hprev + (size_t)r15 * HID;
    for (int k = 0; k < HID; k += 32)
      acc = wmma_bf16(load_fragA(arow, k, half), load_fragB(brow, k, half), acc);
  }
#pragma unroll
  for (int r = 0; r < 8; ++r) proj[r + 8 * half][w * 16 + r15] = acc[r];
  __syncthreads();

  if (tid < 8) {   // per-row master softmax + cumsum
    int row = tid;
    float e[16], mx = -1e30f, s = 0.f, cum = 0.f;
    for (int j = 0; j < 16; ++j) mx = fmaxf(mx, proj[row][128 + j]);
    for (int j = 0; j < 16; ++j) { e[j] = expf(proj[row][128 + j] - mx); s += e[j]; }
    for (int j = 0; j < 16; ++j) { cum += e[j] / s; tfs[row][j] = cum; }
    mx = -1e30f; s = 0.f; cum = 0.f;
    for (int j = 0; j < 16; ++j) mx = fmaxf(mx, proj[row][144 + j]);
    for (int j = 0; j < 16; ++j) { e[j] = expf(proj[row][144 + j] - mx); s += e[j]; }
    for (int j = 0; j < 16; ++j) { cum += e[j] / s; tis[row][j] = 1.f - cum; }
    if (out_cf && blockIdx.x == 0) {
      float sum = 0.f;
      for (int j = 0; j < 16; ++j) {
        out_cf[((size_t)row * TL + t) * 16 + j] = tfs[row][j];
        sum += tfs[row][j];
      }
      out_df[(size_t)row * TL + t] = (float)NCH - sum;
    }
  }
  __syncthreads();

  if (tid < 256) {  // cell update for this block's 32 hidden columns x 8 rows
    int row = tid >> 5, nl = tid & 31, n = n0 + nl;
    float iv = proj[row][0 * 32 + nl], fv = proj[row][1 * 32 + nl];
    float ov = proj[row][2 * 32 + nl], gv = proj[row][3 * 32 + nl];
    int chunk = n >> 6;
    float tf = tfs[row][chunk], ti = tis[row][chunk], wgt = tf * ti;
    float ig = 1.f / (1.f + expf(-iv)), fg = 1.f / (1.f + expf(-fv));
    float og = 1.f / (1.f + expf(-ov)), gg = tanhf(gv);
    float c2 = (fg * wgt + (tf - wgt)) * cstate[(size_t)row * HID + n] +
               (ig * wgt + (ti - wgt)) * gg;
    float h2 = og * tanhf(c2);
    cstate[(size_t)row * HID + n] = c2;
    hnext[(size_t)row * HID + n] = (__bf16)h2;
    if (h_all) h_all[((size_t)row * TL + t) * HID + n] = (__bf16)h2;
  }
}

// ---------------------------------------------------------------------------
// log_prior = log_softmax(h @ Wp + bp)  (tiny: K=5)
// ---------------------------------------------------------------------------
__global__ void k_prior(const __bf16* __restrict__ h_all, const float* __restrict__ Wp,
                        const float* __restrict__ bp, float* __restrict__ lprior) {
  int row = blockIdx.x * blockDim.x + threadIdx.x;
  if (row >= BL) return;
  float z[KMIX];
#pragma unroll
  for (int j = 0; j < KMIX; ++j) z[j] = bp[j];
  const __bf16* h = h_all + (size_t)row * HID;
  for (int k = 0; k < HID; ++k) {
    float hv = (float)h[k];
#pragma unroll
    for (int j = 0; j < KMIX; ++j) z[j] += hv * Wp[k * KMIX + j];
  }
  float mx = z[0];
#pragma unroll
  for (int j = 1; j < KMIX; ++j) mx = fmaxf(mx, z[j]);
  float s = 0.f;
#pragma unroll
  for (int j = 0; j < KMIX; ++j) s += expf(z[j] - mx);
  float lse = mx + logf(s);
#pragma unroll
  for (int j = 0; j < KMIX; ++j) lprior[row * KMIX + j] = z[j] - lse;
}

// ---------------------------------------------------------------------------
// lat = tanh(h @ Wl + bl), stored bf16 in mixture-row order [(bl*5+k)][h]
// ---------------------------------------------------------------------------
__global__ void k_lat(const __bf16* __restrict__ h_all, const __bf16* __restrict__ Wlt,
                      const float* __restrict__ blv, __bf16* __restrict__ lat) {
  int wid = (blockIdx.x * 256 + threadIdx.x) >> 5;
  int lane = threadIdx.x & 31;
  int mt = wid / 320, nt = wid % 320;
  int half = lane >> 4, r15 = lane & 15;
  const __bf16* arow = h_all + (size_t)(mt * 16 + r15) * HID;
  int ncol = nt * 16 + r15;
  const __bf16* brow = Wlt + (size_t)ncol * HID;
  v8f acc;
  float bv = blv[ncol];
#pragma unroll
  for (int i = 0; i < 8; ++i) acc[i] = bv;
  for (int k = 0; k < HID; k += 32)
    acc = wmma_bf16(load_fragA(arow, k, half), load_fragB(brow, k, half), acc);
  int kk = ncol >> 10, hh = ncol & (HID - 1);
#pragma unroll
  for (int r = 0; r < 8; ++r) {
    int row = mt * 16 + r + 8 * half;   // (b,l) flat row
    lat[((size_t)row * KMIX + kk) * HID + hh] = (__bf16)tanhf(acc[r]);
  }
}

// ---------------------------------------------------------------------------
// Big mixture GEMM: logits[r][v] = lat[r] . WoT[v],  M=5120, N=32000, K=1024
// grid = (5120/MBLK) * NSPL blocks, 8 waves. TDM stages the MBLKx1024 A block
// into LDS once; each wave computes a 32x32 tile (2x2 WMMA) per N iteration.
// ---------------------------------------------------------------------------
__global__ __launch_bounds__(256) void k_logits(const __bf16* __restrict__ lat,
                                                const __bf16* __restrict__ Wot,
                                                __bf16* __restrict__ logits) {
  __shared__ __bf16 sA[MBLK][HID];            // 64 KB
  int tid = threadIdx.x, lane = tid & 31, w = tid >> 5;
  int half = lane >> 4, r15 = lane & 15;
  int mb = blockIdx.x / NSPL;                 // 0..159
  int ns = blockIdx.x % NSPL;                 // 0..NSPL-1

  if (w == 0) {
    // Tensor DMA: global (lat + mb*MBLK*HID) -> LDS sA, 2D tile 1024 x MBLK,
    // 2-byte elements, dim0 stride 1024. D# per CDNA5 ISA 8.3/8.4.
    unsigned long long ga =
        (unsigned long long)(const void*)(lat + (size_t)mb * MBLK * HID);
    unsigned ldsoff = (unsigned)(unsigned long long)(void*)&sA[0][0];
    u32x4 g0;
    g0[0] = 1u;                                   // count=1, user descriptor
    g0[1] = ldsoff;                               // lds_addr
    g0[2] = (unsigned)ga;                         // global_addr[31:0]
    g0[3] = (unsigned)((ga >> 32) & 0x01FFFFFFu)  // global_addr[56:32]
            | 0x80000000u;                        // type=2 ("image")
    i32x8 g1;
    g1[0] = 0x00010000;                 // data_size=1 (2 bytes)
    g1[1] = (int)(1024u << 16);         // tensor_dim0[15:0]=1024
    g1[2] = (int)((unsigned)MBLK << 16);// tensor_dim0[31:16]=0, tensor_dim1=MBLK
    g1[3] = (int)(1024u << 16);         // tile_dim0=1024
    g1[4] = MBLK;                       // tile_dim1=MBLK, tile_dim2=0
    g1[5] = 1024;                       // tensor_dim0_stride[31:0]
    g1[6] = 0;                          // stride0 hi, stride1 lo
    g1[7] = 0;
    i32x4 z4;
    z4[0] = 0; z4[1] = 0; z4[2] = 0; z4[3] = 0;
    i32x8 z8;
#pragma unroll
    for (int i = 0; i < 8; ++i) z8[i] = 0;
    __builtin_amdgcn_tensor_load_to_lds(g0, g1, z4, z4, z8, 0);
    __builtin_amdgcn_s_wait_tensorcnt(0);
  }
  __syncthreads();
  asm volatile("" ::: "memory");   // LDS was written by TDM, not visible to IR

  const __bf16* a0 = &sA[r15][0];
  const __bf16* a1 = &sA[16 + r15][0];
  const int iters = (VOC / NSPL) / 256;   // 25 iterations of 256 columns
  for (int it = 0; it < iters; ++it) {
    int nbase = ns * (VOC / NSPL) + it * 256 + w * 32;
    int nc0 = nbase + r15, nc1 = nbase + 16 + r15;
    const __bf16* b0p = Wot + (size_t)nc0 * HID;
    const __bf16* b1p = Wot + (size_t)nc1 * HID;
    v8f acc00, acc01, acc10, acc11;
#pragma unroll
    for (int i = 0; i < 8; ++i) {
      acc00[i] = 0.f; acc01[i] = 0.f; acc10[i] = 0.f; acc11[i] = 0.f;
    }
    for (int k = 0; k < HID; k += 32) {
      __builtin_prefetch(b0p + k + 256, 0, 1);   // global_prefetch_b8
      v16bf af0 = load_fragA(a0, k, half);       // LDS
      v16bf af1 = load_fragA(a1, k, half);       // LDS
      v16bf bf0 = load_fragB(b0p, k, half);      // global (L2-resident Wo)
      v16bf bf1 = load_fragB(b1p, k, half);
      acc00 = wmma_bf16(af0, bf0, acc00);
      acc01 = wmma_bf16(af0, bf1, acc01);
      acc10 = wmma_bf16(af1, bf0, acc10);
      acc11 = wmma_bf16(af1, bf1, acc11);
    }
#pragma unroll
    for (int r = 0; r < 8; ++r) {
      int row0 = mb * MBLK + r + 8 * half;
      int row1 = row0 + 16;
      logits[(size_t)row0 * VOC + nc0] = (__bf16)acc00[r];
      logits[(size_t)row0 * VOC + nc1] = (__bf16)acc01[r];
      logits[(size_t)row1 * VOC + nc0] = (__bf16)acc10[r];
      logits[(size_t)row1 * VOC + nc1] = (__bf16)acc11[r];
    }
  }
}

// ---------------------------------------------------------------------------
// Per-(b,l): mix log_softmax over V, logsumexp over K, NLL at gold, argmax.
// ---------------------------------------------------------------------------
__global__ void k_reduce(const __bf16* __restrict__ logits,
                         const float* __restrict__ lprior, const float* __restrict__ bo,
                         const int* __restrict__ tok, float* __restrict__ nllm,
                         float* __restrict__ maskv, float* __restrict__ pred_out) {
  int bl = blockIdx.x;          // b*128 + l
  int b = bl >> 7, l = bl & 127;
  int tid = threadIdx.x;        // 256
  __shared__ float red[256];
  __shared__ int redi[256];
  __shared__ float ak[KMIX];
  __shared__ float sg;
  int gold = tok[b * TT + (l + 1)];
  const __bf16* base = logits + (size_t)bl * KMIX * VOC;
  float m[KMIX], s[KMIX];
  for (int k = 0; k < KMIX; ++k) {
    float mx = -1e30f;
    for (int v = tid; v < VOC; v += 256)
      mx = fmaxf(mx, (float)base[(size_t)k * VOC + v] + bo[v]);
    red[tid] = mx; __syncthreads();
    for (int o = 128; o > 0; o >>= 1) {
      if (tid < o) red[tid] = fmaxf(red[tid], red[tid + o]);
      __syncthreads();
    }
    m[k] = red[0]; __syncthreads();
  }
  for (int k = 0; k < KMIX; ++k) {
    float ss = 0.f;
    for (int v = tid; v < VOC; v += 256)
      ss += expf((float)base[(size_t)k * VOC + v] + bo[v] - m[k]);
    red[tid] = ss; __syncthreads();
    for (int o = 128; o > 0; o >>= 1) {
      if (tid < o) red[tid] += red[tid + o];
      __syncthreads();
    }
    s[k] = red[0]; __syncthreads();
  }
  if (tid < KMIX) ak[tid] = lprior[bl * KMIX + tid] - m[tid] - logf(s[tid]);
  if (tid == 0) sg = 0.f;
  __syncthreads();
  float bmax = -1e30f; int bidx = 0;
  for (int v = tid; v < VOC; v += 256) {
    float acc = 0.f;
#pragma unroll
    for (int k = 0; k < KMIX; ++k)
      acc += expf(ak[k] + (float)base[(size_t)k * VOC + v] + bo[v]);
    float lp = logf(acc);
    if (lp > bmax || (lp == bmax && v < bidx)) { bmax = lp; bidx = v; }
    if (v == gold) sg = -lp;        // exactly one thread writes
  }
  red[tid] = bmax; redi[tid] = bidx; __syncthreads();
  for (int o = 128; o > 0; o >>= 1) {
    if (tid < o) {
      if (red[tid + o] > red[tid] ||
          (red[tid + o] == red[tid] && redi[tid + o] < redi[tid])) {
        red[tid] = red[tid + o]; redi[tid] = redi[tid + o];
      }
    }
    __syncthreads();
  }
  if (tid == 0) {
    pred_out[bl] = (float)redi[0];
    float msk = (gold != 0) ? 1.f : 0.f;
    nllm[bl] = msk * sg;
    maskv[bl] = msk;
  }
}

__global__ void k_loss(const float* __restrict__ nllm, const float* __restrict__ maskv,
                       float* __restrict__ out) {
  __shared__ float ps[BB];
  int b = threadIdx.x;
  if (b < BB) {
    float sn = 0.f, sm = 0.f;
    for (int l = 0; l < TL; ++l) { sn += nllm[b * TL + l]; sm += maskv[b * TL + l]; }
    ps[b] = sn / (sm + 1e-13f);
  }
  __syncthreads();
  if (b == 0) {
    float a = 0.f;
    for (int i = 0; i < BB; ++i) a += ps[i];
    out[0] = a / (float)BB;
  }
}

// ---------------------------------------------------------------------------
// Host-side orchestration
// ---------------------------------------------------------------------------
extern "C" void kernel_launch(void* const* d_in, const int* in_sizes, int n_in,
                              void* d_out, int out_size, void* d_ws, size_t ws_size,
                              hipStream_t stream) {
  const int*   tok = (const int*)  d_in[0];
  const float* emb = (const float*)d_in[1];
  const float* Wx0 = (const float*)d_in[2];
  const float* Wh0 = (const float*)d_in[3];
  const float* b0  = (const float*)d_in[4];
  const float* Wx1 = (const float*)d_in[5];
  const float* Wh1 = (const float*)d_in[6];
  const float* b1  = (const float*)d_in[7];
  const float* Wx2 = (const float*)d_in[8];
  const float* Wh2 = (const float*)d_in[9];
  const float* b2  = (const float*)d_in[10];
  const float* Wp  = (const float*)d_in[11];
  const float* bp  = (const float*)d_in[12];
  const float* Wl  = (const float*)d_in[13];
  const float* blv = (const float*)d_in[14];
  const float* Wo  = (const float*)d_in[15];
  const float* bo  = (const float*)d_in[16];
  float* out = (float*)d_out;

  char* p = (char*)d_ws;
  auto carve = [&](size_t bytes) -> void* {
    void* r = (void*)p;
    p += (bytes + 255) & ~(size_t)255;
    return r;
  };
  __bf16* emb_bf = (__bf16*)carve((size_t)VOC * EMB * 2);
  __bf16* Wx0t   = (__bf16*)carve((size_t)GATE * EMB * 2);
  __bf16* Wh0t   = (__bf16*)carve((size_t)GATE * HID * 2);
  __bf16* Wx1t   = (__bf16*)carve((size_t)GATE * HID * 2);
  __bf16* Wh1t   = (__bf16*)carve((size_t)GATE * HID * 2);
  __bf16* Wx2t   = (__bf16*)carve((size_t)GATE * HID * 2);
  __bf16* Wh2t   = (__bf16*)carve((size_t)GATE * HID * 2);
  __bf16* Wlt    = (__bf16*)carve((size_t)KMIX * HID * HID * 2);
  __bf16* Wot    = (__bf16*)carve((size_t)VOC * HID * 2);
  float*  Xp     = (float*) carve((size_t)BL * GATE * 4);
  __bf16* hbuf   = (__bf16*)carve((size_t)3 * 2 * 16 * HID * 2);
  float*  cbuf   = (float*) carve((size_t)3 * 8 * HID * 4);
  __bf16* h_all  = (__bf16*)carve((size_t)BL * HID * 2);
  float*  lprior = (float*) carve((size_t)BL * KMIX * 4);
  __bf16* lat    = (__bf16*)carve((size_t)MROW * HID * 2);
  __bf16* logits = (__bf16*)carve((size_t)MROW * VOC * 2);
  float*  nllm   = (float*) carve((size_t)BL * 4);
  float*  maskv  = (float*) carve((size_t)BL * 4);
  (void)ws_size; (void)in_sizes; (void)n_in; (void)out_size;

  // --- prep: bf16 conversion + weight transposes ---
  {
    size_t n = (size_t)VOC * EMB;
    k_cvt<<<(unsigned)((n + 255) / 256), 256, 0, stream>>>(emb, emb_bf, n);
  }
  dim3 tb(32, 8);
  k_transpose<<<dim3(GATE / 32, EMB / 32), tb, 0, stream>>>(Wx0, Wx0t, EMB, GATE);
  k_transpose<<<dim3(GATE / 32, HID / 32), tb, 0, stream>>>(Wh0, Wh0t, HID, GATE);
  k_transpose<<<dim3(GATE / 32, HID / 32), tb, 0, stream>>>(Wx1, Wx1t, HID, GATE);
  k_transpose<<<dim3(GATE / 32, HID / 32), tb, 0, stream>>>(Wh1, Wh1t, HID, GATE);
  k_transpose<<<dim3(GATE / 32, HID / 32), tb, 0, stream>>>(Wx2, Wx2t, HID, GATE);
  k_transpose<<<dim3(GATE / 32, HID / 32), tb, 0, stream>>>(Wh2, Wh2t, HID, GATE);
  k_transpose<<<dim3(KMIX * HID / 32, HID / 32), tb, 0, stream>>>(Wl, Wlt, HID, KMIX * HID);
  k_transpose<<<dim3(VOC / 32, HID / 32), tb, 0, stream>>>(Wo, Wot, HID, VOC);
  k_init<<<384, 256, 0, stream>>>(hbuf, cbuf);

  // --- parallel precompute of x @ Wx0 + b0 ---
  k_xproj<<<(64 * 258) / 8, 256, 0, stream>>>(tok, emb_bf, Wx0t, b0, Xp);

  // --- sequential recurrence: 128 steps x 3 layers ---
  const size_t HB = (size_t)16 * HID;        // elements per h buffer
  for (int t = 0; t < TL; ++t) {
    int rb = t & 1, wb = rb ^ 1;
    __bf16* h0r = hbuf + (0 * 2 + rb) * HB;
    __bf16* h0w = hbuf + (0 * 2 + wb) * HB;
    __bf16* h1r = hbuf + (1 * 2 + rb) * HB;
    __bf16* h1w = hbuf + (1 * 2 + wb) * HB;
    __bf16* h2r = hbuf + (2 * 2 + rb) * HB;
    __bf16* h2w = hbuf + (2 * 2 + wb) * HB;
    // layer 0: proj = Xp[t] + h0 @ Wh0
    k_lstm_step<<<32, 320, 0, stream>>>(t, Xp, (const __bf16*)nullptr, Wh0t,
                                        (const float*)nullptr, (const __bf16*)nullptr,
                                        h0r, h0w, cbuf + 0 * 8 * HID,
                                        (__bf16*)nullptr, (float*)nullptr,
                                        (float*)nullptr);
    // layer 1: proj = b1 + h0 @ Wx1 + h1 @ Wh1 ; emits cf/df
    k_lstm_step<<<32, 320, 0, stream>>>(t, (const float*)nullptr, Wx1t, Wh1t, b1,
                                        h0w, h1r, h1w, cbuf + 1 * 8 * HID,
                                        (__bf16*)nullptr,
                                        out + 1, out + 1 + BB * TL);
    // layer 2: proj = b2 + h1 @ Wx2 + h2 @ Wh2 ; emits h_all
    k_lstm_step<<<32, 320, 0, stream>>>(t, (const float*)nullptr, Wx2t, Wh2t, b2,
                                        h1w, h2r, h2w, cbuf + 2 * 8 * HID,
                                        h_all, (float*)nullptr, (float*)nullptr);
  }

  // --- output heads ---
  k_prior<<<(BL + 255) / 256, 256, 0, stream>>>(h_all, Wp, bp, lprior);
  k_lat<<<(64 * 320) / 8, 256, 0, stream>>>(h_all, Wlt, blv, lat);
  k_logits<<<(MROW / MBLK) * NSPL, 256, 0, stream>>>(lat, Wot, logits);
  k_reduce<<<BL, 256, 0, stream>>>(logits, lprior, bo, tok, nllm, maskv,
                                   out + 1 + BB * TL + BB * TL * NCH);
  k_loss<<<1, 32, 0, stream>>>(nllm, maskv, out);
}
